// LSTMAEDecoderReverse_11441792876599
// MI455X (gfx1250) — matrix operand order repaired and checked by reading, not compiled, not run on hardware
//
#include <hip/hip_runtime.h>
#include <hip/hip_bf16.h>
#include <math.h>

// Problem constants (reference: B,ENC_H,HID_H,D,T = 128,512,512,256,256)
#define B_    128
#define DX    256          // decoder input / output feature dim
#define H     512          // hidden dim (both layers)
#define G4    (4*H)        // 2048 gate rows
#define T_    256
#define NWG   16           // one workgroup cluster (<=16 WGs on one SE)
#define NTHR  512          // 16 waves per WG
#define WPB   (NTHR/32)

typedef __attribute__((ext_vector_type(16))) _Float16 v16h;
typedef __attribute__((ext_vector_type(8)))  float    v8f;
typedef __attribute__((ext_vector_type(4)))  float    f32x4;
typedef __attribute__((ext_vector_type(4)))  unsigned int u32x4;
typedef __attribute__((ext_vector_type(8)))  int      i32x8;
typedef __attribute__((ext_vector_type(4)))  int      i32x4;

union V16 { v16h v; f32x4 q[2]; };

struct WS {
  _Float16* wih0; _Float16* whh0; _Float16* wih1; _Float16* whh1; _Float16* wlin;
  float* b0; float* b1; float* bl;
  _Float16* h0[2]; _Float16* h1[2]; _Float16* x[2];
  float* c0; float* c1;
  unsigned* arrive; unsigned* release;
};

// ---------------------------------------------------------------------------
// one-time setup: f32 -> f16 weight conversion, fused biases, initial states
// ---------------------------------------------------------------------------
__global__ void lstm_init(const float* Wih0, const float* Whh0,
                          const float* Wih1, const float* Whh1,
                          const float* Wlin,
                          const float* bih0, const float* bhh0,
                          const float* bih1, const float* bhh1,
                          const float* blin,
                          const float* hid, WS ws)
{
  const long i0 = (long)blockIdx.x * blockDim.x + threadIdx.x;
  const long st = (long)gridDim.x * blockDim.x;
  for (long i = i0; i < (long)G4*DX; i += st) ws.wih0[i] = (_Float16)Wih0[i];
  for (long i = i0; i < (long)G4*H;  i += st) ws.whh0[i] = (_Float16)Whh0[i];
  for (long i = i0; i < (long)G4*H;  i += st) ws.wih1[i] = (_Float16)Wih1[i];
  for (long i = i0; i < (long)G4*H;  i += st) ws.whh1[i] = (_Float16)Whh1[i];
  for (long i = i0; i < (long)DX*H;  i += st) ws.wlin[i] = (_Float16)Wlin[i];
  for (long i = i0; i < G4; i += st) { ws.b0[i] = bih0[i] + bhh0[i];
                                       ws.b1[i] = bih1[i] + bhh1[i]; }
  for (long i = i0; i < DX; i += st) ws.bl[i] = blin[i];
  // hidden = initial_hidden[::-1] -> h0 = initial_hidden[1], h1 = initial_hidden[0]
  for (long i = i0; i < (long)B_*H; i += st) {
    ws.h0[0][i] = (_Float16)hid[(long)B_*H + i];
    ws.h1[0][i] = (_Float16)hid[i];
    ws.c0[i] = 0.f; ws.c1[i] = 0.f;
  }
  if (i0 == 0) { *ws.arrive = 0u; *ws.release = 0u; }
}

// ---------------------------------------------------------------------------
// helpers
// ---------------------------------------------------------------------------
__device__ __forceinline__ float sigm(float x)   { return 1.f / (1.f + __expf(-x)); }
__device__ __forceinline__ float tanhf_(float x) { float e = __expf(2.f * x); return 1.f - 2.f / (e + 1.f); }

// ---------------------------------------------------------------------------
// Tensor Data Mover: 1-row tile, data_size = 8B, global -> LDS.
// D# built per CDNA5 ISA 8.3/8.4 (group0: count/lds_addr/global_addr/type,
// group1: data_size + tensor_dim0/1 + tile_dim0 + stride0).
// nbytes must be a multiple of 8 and <= 512KB (tile_dim0 <= 65535).
// ---------------------------------------------------------------------------
__device__ __forceinline__ void tdm_load(void* ldsDst, const void* gsrc, unsigned nbytes)
{
  const unsigned long long ga = (unsigned long long)gsrc;
  const unsigned lds_off = (unsigned)(unsigned long long)(size_t)ldsDst; // flat low 32 = LDS byte offset
  const unsigned n8 = nbytes >> 3;
  u32x4 g0;
  g0[0] = 1u;                                               // count=1 (valid user D#)
  g0[1] = lds_off;                                          // lds_addr (bytes)
  g0[2] = (unsigned)(ga & 0xFFFFFFFFu);                     // global_addr[31:0]
  g0[3] = (unsigned)((ga >> 32) & 0x1FFFFFFu) | (2u << 30); // global_addr[56:32] | type=2
  i32x8 g1;
  g1[0] = (int)(3u << 16);                                  // data_size=3 (8B), no multicast
  g1[1] = (int)((n8 & 0xFFFFu) << 16);                      // tensor_dim0[15:0]
  g1[2] = (int)(((n8 >> 16) & 0xFFFFu) | (1u << 16));       // tensor_dim0[31:16], tensor_dim1=1
  g1[3] = (int)((n8 & 0xFFFFu) << 16);                      // tile_dim0 = n8
  g1[4] = 0;                                                // tile_dim1/2 unused
  g1[5] = (int)n8;                                          // tensor_dim0_stride[31:0]
  g1[6] = 0;
  g1[7] = 0;
  i32x4 z4 = {0, 0, 0, 0};
#if __clang_major__ >= 23
  i32x8 z8 = {0, 0, 0, 0, 0, 0, 0, 0};
  __builtin_amdgcn_tensor_load_to_lds(g0, g1, z4, z4, z8, 0);
#else
  __builtin_amdgcn_tensor_load_to_lds(g0, g1, z4, z4, 0);
#endif
}

// 16x16 f32 tile  +=  A(16xK from LDS) * B(Kx16 = rows of row-major W from L2)
// A lane layout (16-bit A 16x32): lane m<16: K {0..7,16..23}; lanes 16-31: +8
// B lane layout (16-bit B 32x16): lanes 0-15 = cols, K {0..7,16..23}; lanes 16-31: +8
__device__ __forceinline__ v8f mm_tile(const _Float16* __restrict__ aRow, int K,
                                       const _Float16* __restrict__ bRow,
                                       v8f acc, int lh)
{
  const int a1 = 8*lh, a2 = 16 + 8*lh, b1 = 8*lh;
  __builtin_prefetch(bRow, 0, 0);
#pragma unroll 4
  for (int k = 0; k < K; k += 32) {
    V16 a, b;
    a.q[0] = *(const f32x4*)(aRow + k + a1);
    a.q[1] = *(const f32x4*)(aRow + k + a2);
    b.q[0] = *(const f32x4*)(bRow + k + b1);
    b.q[1] = *(const f32x4*)(bRow + k + b1 + 16);
    acc = __builtin_amdgcn_wmma_f32_16x16x32_f16(false, a.v, false, b.v,
                                                 (short)0, acc, false, false);
  }
  return acc;
}

// cluster-wide barrier: hardware cluster barrier (fast path / NOP outside a
// cluster) + sense-reversing L2 atomic barrier (always correct for a
// persistent 16-WG grid). Release/acquire via __threadfence.
__device__ __forceinline__ void grid_sync(unsigned* arrive, unsigned* release, unsigned e)
{
  __syncthreads();
  __builtin_amdgcn_s_cluster_barrier();     // s_barrier_signal/wait -3
  if (threadIdx.x == 0) {
    __threadfence();
    unsigned prev = __hip_atomic_fetch_add(arrive, 1u, __ATOMIC_ACQ_REL,
                                           __HIP_MEMORY_SCOPE_AGENT);
    if (prev == NWG - 1) {
      __hip_atomic_store(arrive, 0u, __ATOMIC_RELAXED, __HIP_MEMORY_SCOPE_AGENT);
      __hip_atomic_store(release, e, __ATOMIC_RELEASE, __HIP_MEMORY_SCOPE_AGENT);
    } else {
      while (__hip_atomic_load(release, __ATOMIC_ACQUIRE,
                               __HIP_MEMORY_SCOPE_AGENT) < e)
        __builtin_amdgcn_s_sleep(1);
    }
    __threadfence();
  }
  __syncthreads();
}

// one LSTM cell layer for one wave's (mt,nt) tile across all 4 gates:
// gates = X@Wx^T + Hs@Wh^T + bias; fused sigmoid/tanh; in-place c update.
__device__ __forceinline__ void cell_phase(const _Float16* ldsX, int Kx,
                                           const _Float16* ldsH,
                                           const _Float16* __restrict__ Wx,
                                           const _Float16* __restrict__ Wh,
                                           const float* __restrict__ bias,
                                           float* __restrict__ c,
                                           _Float16* __restrict__ hout,
                                           int wave, int lm, int lh)
{
  const int mt = wave >> 5, nt = wave & 31;      // 8 x 32 = 256 jobs = 256 waves
  const int m0 = mt * 16, n0 = nt * 16;
  const _Float16* aX = ldsX + (m0 + lm) * Kx;
  const _Float16* aH = ldsH + (m0 + lm) * H;
  v8f acc[4];
#pragma unroll
  for (int g = 0; g < 4; ++g) {
    const int wr = g * H + n0 + lm;              // weight row = gate column
    const float bb = bias[g * H + n0 + lm];
    v8f a = {bb, bb, bb, bb, bb, bb, bb, bb};
    a = mm_tile(aX, Kx, Wx + (long)wr * Kx, a, lh);
    a = mm_tile(aH, H,  Wh + (long)wr * H,  a, lh);
    acc[g] = a;
  }
#pragma unroll
  for (int r = 0; r < 8; ++r) {
    const int m = m0 + r + 8 * lh;               // D-fragment: M = r (+8 hi half)
    const int n = n0 + lm;                       //             N = lane 0..15
    const float cv = c[m * H + n];
    const float ig = sigm(acc[0][r]);
    const float fg = sigm(acc[1][r]);
    const float gg = tanhf_(acc[2][r]);
    const float og = sigm(acc[3][r]);
    const float cn = fg * cv + ig * gg;
    c[m * H + n] = cn;
    hout[m * H + n] = (_Float16)(og * tanhf_(cn));
  }
}

// out = h1 @ Wlin^T + blin ; writes f32 to d_out slice and f16 feedback x
__device__ __forceinline__ void out_phase(const _Float16* ldsH,
                                          const _Float16* __restrict__ Wl,
                                          const float* __restrict__ bl,
                                          float* __restrict__ outT,
                                          _Float16* __restrict__ xout,
                                          int wave, int lm, int lh)
{
  if (wave >= 128) return;                       // 8 x 16 = 128 jobs
  const int mt = wave >> 4, nt = wave & 15;
  const int m0 = mt * 16, n0 = nt * 16;
  const _Float16* aH = ldsH + (m0 + lm) * H;
  const float bb = bl[n0 + lm];
  v8f acc = {bb, bb, bb, bb, bb, bb, bb, bb};
  acc = mm_tile(aH, H, Wl + (long)(n0 + lm) * H, acc, lh);
#pragma unroll
  for (int r = 0; r < 8; ++r) {
    const int m = m0 + r + 8 * lh;
    const int n = n0 + lm;
    outT[m * DX + n] = acc[r];
    xout[m * DX + n] = (_Float16)acc[r];
  }
}

// ---------------------------------------------------------------------------
// persistent cluster kernel: whole T=256 sequence in one launch.
// LDS (320 KB): P = h0 image (128K), Q = h1 image (128K), RX = x image (64K).
// Steady-state invariant: at phase-A entry P holds current h0, Q holds
// current h1 (they were the staged X-operands of the previous phases), so
// each phase needs exactly ONE TDM stage on the critical path.
// ---------------------------------------------------------------------------
__global__ __launch_bounds__(NTHR) void lstm_main(WS ws, float* __restrict__ out)
{
  extern __shared__ _Float16 lds[];
  _Float16* P  = lds;                  // 65536 halves (128 KB) : h0
  _Float16* Q  = lds + 65536;          // 65536 halves (128 KB) : h1
  _Float16* RX = lds + 131072;         // 32768 halves ( 64 KB) : x
  const int wave = blockIdx.x * WPB + (threadIdx.x >> 5);
  const bool issuer = (threadIdx.x >> 5) == 0;   // wave 0 drives the TDM
  const int lm = threadIdx.x & 15;
  const int lh = (threadIdx.x >> 4) & 1;
  unsigned ep = 0;

  // ---- t = 0: DMA initial states into LDS, out0 = h1_init @ Wlin^T + blin
  if (issuer) {
    tdm_load(P, ws.h0[0], B_ * H * 2);
    tdm_load(Q, ws.h1[0], B_ * H * 2);
    __builtin_amdgcn_s_wait_tensorcnt(0);
  }
  __syncthreads();
  out_phase(Q, ws.wlin, ws.bl, out + (long)(T_ - 1) * B_ * DX, ws.x[0],
            wave, lm, lh);
  grid_sync(ws.arrive, ws.release, ++ep);

  for (int t = 1; t < T_; ++t) {
    const int rd = (t - 1) & 1, wr = t & 1;

    // ---- layer 0: gates0 = x@Wih0^T + h0@Whh0^T + b0   (h0 already in P)
    if (issuer) {
      tdm_load(RX, ws.x[rd], B_ * DX * 2);
      __builtin_amdgcn_s_wait_tensorcnt(0);
    }
    __syncthreads();
    cell_phase(RX, DX, P, ws.wih0, ws.whh0, ws.b0,
               ws.c0, ws.h0[wr], wave, lm, lh);
    grid_sync(ws.arrive, ws.release, ++ep);

    // ---- layer 1: gates1 = h0n@Wih1^T + h1@Whh1^T + b1 (h1 already in Q)
    if (issuer) {
      tdm_load(P, ws.h0[wr], B_ * H * 2);      // P becomes h0n == next h0
      __builtin_amdgcn_s_wait_tensorcnt(0);
    }
    __syncthreads();
    cell_phase(P, H, Q, ws.wih1, ws.whh1, ws.b1,
               ws.c1, ws.h1[wr], wave, lm, lh);
    grid_sync(ws.arrive, ws.release, ++ep);

    // ---- head: out_t = h1n@Wlin^T + blin ; result[T-1-t] = out_t
    if (issuer) {
      tdm_load(Q, ws.h1[wr], B_ * H * 2);      // Q becomes h1n == next h1
      __builtin_amdgcn_s_wait_tensorcnt(0);
    }
    __syncthreads();
    out_phase(Q, ws.wlin, ws.bl, out + (long)(T_ - 1 - t) * B_ * DX,
              ws.x[wr], wave, lm, lh);
    grid_sync(ws.arrive, ws.release, ++ep);
  }
}

// ---------------------------------------------------------------------------
extern "C" void kernel_launch(void* const* d_in, const int* in_sizes, int n_in,
                              void* d_out, int out_size, void* d_ws, size_t ws_size,
                              hipStream_t stream)
{
  (void)in_sizes; (void)n_in; (void)out_size; (void)ws_size;
  const float* hid  = (const float*)d_in[0];
  const float* Wih0 = (const float*)d_in[1];
  const float* Whh0 = (const float*)d_in[2];
  const float* bih0 = (const float*)d_in[3];
  const float* bhh0 = (const float*)d_in[4];
  const float* Wih1 = (const float*)d_in[5];
  const float* Whh1 = (const float*)d_in[6];
  const float* bih1 = (const float*)d_in[7];
  const float* bhh1 = (const float*)d_in[8];
  const float* Wlin = (const float*)d_in[9];
  const float* blin = (const float*)d_in[10];

  // carve the workspace (all offsets 256B-aligned)
  char* p = (char*)d_ws;
  size_t off = 0;
  auto take = [&](size_t bytes) {
    char* r = p + off;
    off += (bytes + 255) & ~(size_t)255;
    return r;
  };
  WS ws;
  ws.wih0 = (_Float16*)take((size_t)G4 * DX * 2);
  ws.whh0 = (_Float16*)take((size_t)G4 * H * 2);
  ws.wih1 = (_Float16*)take((size_t)G4 * H * 2);
  ws.whh1 = (_Float16*)take((size_t)G4 * H * 2);
  ws.wlin = (_Float16*)take((size_t)DX * H * 2);
  ws.b0   = (float*)take((size_t)G4 * 4);
  ws.b1   = (float*)take((size_t)G4 * 4);
  ws.bl   = (float*)take((size_t)DX * 4);
  ws.h0[0] = (_Float16*)take((size_t)B_ * H * 2);
  ws.h0[1] = (_Float16*)take((size_t)B_ * H * 2);
  ws.h1[0] = (_Float16*)take((size_t)B_ * H * 2);
  ws.h1[1] = (_Float16*)take((size_t)B_ * H * 2);
  ws.x[0]  = (_Float16*)take((size_t)B_ * DX * 2);
  ws.x[1]  = (_Float16*)take((size_t)B_ * DX * 2);
  ws.c0    = (float*)take((size_t)B_ * H * 4);
  ws.c1    = (float*)take((size_t)B_ * H * 4);
  ws.arrive  = (unsigned*)take(256);
  ws.release = ws.arrive + 16;

  hipFuncSetAttribute(reinterpret_cast<const void*>(lstm_main),
                      hipFuncAttributeMaxDynamicSharedMemorySize, 327680);

  lstm_init<<<1024, 256, 0, stream>>>(Wih0, Whh0, Wih1, Whh1, Wlin,
                                      bih0, bhh0, bih1, bhh1, blin, hid, ws);
  lstm_main<<<NWG, NTHR, 327680, stream>>>(ws, (float*)d_out);
}
